// MultiHeadAttention_59863254172224
// MI455X (gfx1250) — matrix operand order repaired and checked by reading, not compiled, not run on hardware
//
#include <hip/hip_runtime.h>
#include <math.h>

// ======================================================================
// MI455X (gfx1250) multi-head attention, bf16 WMMA flash-attention.
// B=2, N=2048, D=1024, H=16, Hd=64.  ~69 GFLOP, ~100MB HBM traffic.
// v4: - flash key tile 32 -> 64 (halves softmax-reduction + barrier
//       overhead per key; 18 WMMAs per iteration).
//     - projection GEMMs stage the shared W tile to LDS via
//       global_load_async_to_lds_b128 (8x less redundant vmem traffic).
// ======================================================================

typedef __attribute__((ext_vector_type(16))) __bf16 bf16x16;
typedef __attribute__((ext_vector_type(8)))  __bf16 bf16x8;
typedef __attribute__((ext_vector_type(8)))  float  f32x8;

union BF16x16 { bf16x16 v; bf16x8 h[2]; };

static constexpr int NSEQ   = 2048;
static constexpr int DMODEL = 1024;
static constexpr int NH     = 16;
static constexpr int HD     = 64;
static constexpr int MTOT   = 2 * NSEQ;   // B*N = 4096

__device__ __forceinline__ f32x8 wmma_bf16(bf16x16 a, bf16x16 b, f32x8 c) {
  return __builtin_amdgcn_wmma_f32_16x16x32_bf16(false, a, false, b, (short)0, c, false, false);
}

__device__ __forceinline__ float redmax16(float v) {
  v = fmaxf(v, __shfl_xor(v, 1, 16));
  v = fmaxf(v, __shfl_xor(v, 2, 16));
  v = fmaxf(v, __shfl_xor(v, 4, 16));
  v = fmaxf(v, __shfl_xor(v, 8, 16));
  return v;
}

// ---- CDNA5 async global->LDS copy (ASYNCcnt-tracked) -----------------
#if __has_builtin(__builtin_amdgcn_global_load_async_to_lds_b128)
#define ATHENA_HAS_ASYNC_LDS 1
#else
#define ATHENA_HAS_ASYNC_LDS 0
#endif

typedef int v4i_b128 __attribute__((vector_size(4 * sizeof(int))));
typedef __attribute__((address_space(1))) v4i_b128* gas_v4i;
typedef __attribute__((address_space(3))) v4i_b128* las_v4i;

__device__ __forceinline__ void async_copy_b128(const void* g, void* l) {
#if ATHENA_HAS_ASYNC_LDS
  __builtin_amdgcn_global_load_async_to_lds_b128((gas_v4i)g, (las_v4i)l, 0, 0);
#else
  *(bf16x8*)l = *(const bf16x8*)g;
#endif
}

__device__ __forceinline__ void wait_async_all() {
#if ATHENA_HAS_ASYNC_LDS
#if __has_builtin(__builtin_amdgcn_s_wait_asynccnt)
  __builtin_amdgcn_s_wait_asynccnt(0);
#else
  asm volatile("s_wait_asynccnt 0" ::: "memory");
#endif
#endif
}

__global__ void cvt_f32_to_bf16(const float* __restrict__ in,
                                __bf16* __restrict__ out, int n) {
  int i = blockIdx.x * blockDim.x + threadIdx.x;
  if (i < n) out[i] = (__bf16)in[i];
}

// ----------------------------------------------------------------------
// GEMM: Y[m,e] = (sum_k X[m,k]*W[e,k] + bias[e]) * oscale
// 256 thr = 8 waves; tile 128(M) x 64(E); wave = 16x64 strip.
// W tile (64 cols x 64 k = 8KB) is shared by all 8 waves -> staged to LDS
// with double-buffered async copies; A fragments are per-wave global b128.
// MODE 0: bf16 out [B,H,N,Hd]; MODE 1: bf16 out [B,H,Hd,N]; MODE 2: f32 [m,e].
// ----------------------------------------------------------------------
template <int MODE>
__global__ __launch_bounds__(256) void proj_gemm(const __bf16* __restrict__ X,
                                                 const __bf16* __restrict__ W,
                                                 const float*  __restrict__ bias,
                                                 void* __restrict__ outv,
                                                 float oscale) {
  __shared__ __bf16 ldsW[2][64][64 + 8];   // 2 x 9 KB, +16B row pad

  const int tid   = threadIdx.x;
  const int wave  = tid >> 5;
  const int lane  = tid & 31;
  const int lrow  = lane & 15;
  const int lhalf = lane >> 4;

  const int mblk = blockIdx.x & 31;   // 4096/128
  const int nblk = blockIdx.x >> 5;   // 1024/64
  const int m0 = mblk * 128 + wave * 16;
  const int n0 = nblk * 64;

  const int srow = tid >> 2;          // 0..63: W column (output channel)
  const int scol = (tid & 3) * 16;    // 0,16,32,48: k offset

  auto stageW = [&](int buf, int k0) {
    const __bf16* g = W + (size_t)(n0 + srow) * DMODEL + k0 + scol;
    async_copy_b128(g,     &ldsW[buf][srow][scol]);
    async_copy_b128(g + 8, &ldsW[buf][srow][scol + 8]);
  };

  f32x8 zero = {0, 0, 0, 0, 0, 0, 0, 0};
  f32x8 acc[4];
#pragma unroll
  for (int t = 0; t < 4; ++t) acc[t] = zero;

  const __bf16* aptr = X + (size_t)(m0 + lrow) * DMODEL;

  stageW(0, 0);

  for (int kb = 0; kb < DMODEL / 64; ++kb) {
    const int cur = kb & 1;
    wait_async_all();
    __syncthreads();
    if (kb + 1 < DMODEL / 64) stageW(cur ^ 1, (kb + 1) * 64);

#pragma unroll
    for (int s = 0; s < 2; ++s) {
      const int k0 = kb * 64 + s * 32;
      BF16x16 af;   // A 16x32: K run = lhalf*16+0..15
      af.h[0] = *(const bf16x8*)(aptr + k0 + lhalf * 16);
      af.h[1] = *(const bf16x8*)(aptr + k0 + lhalf * 16 + 8);
#pragma unroll
      for (int nt = 0; nt < 4; ++nt) {
        BF16x16 bfr;  // B 32x16: lane=col, K run = lhalf*16+0..15
        const __bf16* p = &ldsW[cur][nt * 16 + lrow][s * 32 + lhalf * 16];
        bfr.h[0] = *(const bf16x8*)(p);
        bfr.h[1] = *(const bf16x8*)(p + 8);
        acc[nt] = wmma_bf16(af.v, bfr.v, acc[nt]);
      }
    }
  }

#pragma unroll
  for (int nt = 0; nt < 4; ++nt) {
    const int ge = n0 + nt * 16 + lrow;
    const float bv = bias[ge];
#pragma unroll
    for (int r = 0; r < 8; ++r) {
      const int gm = m0 + r + 8 * lhalf;
      const float v = (acc[nt][r] + bv) * oscale;
      if constexpr (MODE == 2) {
        ((float*)outv)[(size_t)gm * DMODEL + ge] = v;
      } else {
        const int b = gm >> 11, n = gm & (NSEQ - 1);
        const int h = ge >> 6,  hd = ge & 63;
        __bf16* o = (__bf16*)outv;
        size_t idx;
        if constexpr (MODE == 0)
          idx = (((size_t)(b * NH + h)) * NSEQ + n) * HD + hd;
        else
          idx = (((size_t)(b * NH + h)) * HD + hd) * NSEQ + n;
        o[idx] = (__bf16)v;
      }
    }
  }
}

// ----------------------------------------------------------------------
// Flash attention.  512 blocks = B*H*(N/128); 8 waves/block, wave owns a
// 16-row Q strip.  64-key tiles: K(64x64) and Vt(64x64) staged to LDS by
// async copies; 8 score WMMAs, online softmax (row-max via shfl, row-sum
// via 2 WMMAs x ones), 8 PV WMMAs per tile.
// ----------------------------------------------------------------------
__global__ __launch_bounds__(256) void flash_attn(const __bf16* __restrict__ Q,
                                                  const __bf16* __restrict__ K,
                                                  const __bf16* __restrict__ Vt,
                                                  __bf16* __restrict__ attn) {
  __shared__ __bf16 ldsK[2][64][HD + 8];   // 2 x 9 KB  [key][d]
  __shared__ __bf16 ldsV[2][HD][64 + 8];   // 2 x 9 KB  [d][key]
  __shared__ __bf16 ldsP[8][16][64 + 8];   // 18 KB     per-wave P tile

  const int tid   = threadIdx.x;
  const int wave  = tid >> 5;
  const int lane  = tid & 31;
  const int lrow  = lane & 15;
  const int lhalf = lane >> 4;

  const int bh = blockIdx.x >> 4;                  // b*H + h
  const int q0 = (blockIdx.x & 15) * 128 + wave * 16;

  const __bf16* Qb = Q  + (size_t)bh * NSEQ * HD;
  const __bf16* Kb = K  + (size_t)bh * NSEQ * HD;
  const __bf16* Vb = Vt + (size_t)bh * HD * NSEQ;

  // Staging: 64x64 tiles, 16 elts/thread (2 x b128).
  const int srow = tid >> 2;          // 0..63
  const int scol = (tid & 3) * 16;    // 0,16,32,48

  auto stage = [&](int buf, int k0) {
    const __bf16* gk = Kb + (size_t)(k0 + srow) * HD + scol;
    async_copy_b128(gk,     &ldsK[buf][srow][scol]);
    async_copy_b128(gk + 8, &ldsK[buf][srow][scol + 8]);
    const __bf16* gv = Vb + (size_t)srow * NSEQ + k0 + scol;
    async_copy_b128(gv,     &ldsV[buf][srow][scol]);
    async_copy_b128(gv + 8, &ldsV[buf][srow][scol + 8]);
  };

  // Preload Q strip as two A-fragments (d = 0..31, 32..63).
  BF16x16 qf[2];
#pragma unroll
  for (int s = 0; s < 2; ++s) {
    const __bf16* p = Qb + (size_t)(q0 + lrow) * HD + s * 32 + lhalf * 16;
    qf[s].h[0] = *(const bf16x8*)(p);
    qf[s].h[1] = *(const bf16x8*)(p + 8);
  }

  // All-ones B fragment for WMMA row-sum.
  BF16x16 onesf;
#pragma unroll
  for (int i = 0; i < 16; ++i) onesf.v[i] = (__bf16)1.0f;

  f32x8 zero = {0, 0, 0, 0, 0, 0, 0, 0};
  f32x8 acc[4];
#pragma unroll
  for (int t = 0; t < 4; ++t) acc[t] = zero;

  float m_i[8], l_i[8];
#pragma unroll
  for (int r = 0; r < 8; ++r) { m_i[r] = -1e30f; l_i[r] = 0.0f; }

  stage(0, 0);

  for (int it = 0; it < NSEQ / 64; ++it) {
    const int cur = it & 1;
    wait_async_all();     // my copies for buffer `cur` are done
    __syncthreads();      // everyone's copies done + old buffer free
    if (it + 1 < NSEQ / 64) stage(cur ^ 1, (it + 1) * 64);

    // ---- scores: S = Q' * K^T  (four 16x16 tiles over 64 keys) ----
    f32x8 sc[4];
#pragma unroll
    for (int t = 0; t < 4; ++t) {
      sc[t] = zero;
#pragma unroll
      for (int s = 0; s < 2; ++s) {
        BF16x16 kf;  // B[d,key]: contiguous along d in the K tile row
        const __bf16* p = &ldsK[cur][t * 16 + lrow][s * 32 + lhalf * 16];
        kf.h[0] = *(const bf16x8*)(p);
        kf.h[1] = *(const bf16x8*)(p + 8);
        sc[t] = wmma_bf16(qf[s].v, kf.v, sc[t]);
      }
    }

    // ---- online softmax: row-max via one 16-lane shfl butterfly ----
    float pv[4][8], esc[8];
#pragma unroll
    for (int r = 0; r < 8; ++r) {
      const float mlocal = fmaxf(fmaxf(sc[0][r], sc[1][r]),
                                 fmaxf(sc[2][r], sc[3][r]));
      const float mx = redmax16(mlocal);
      const float mn = fmaxf(m_i[r], mx);
      esc[r] = __expf(m_i[r] - mn);
#pragma unroll
      for (int t = 0; t < 4; ++t) pv[t][r] = __expf(sc[t][r] - mn);
      m_i[r] = mn;
    }

    // ---- P: C-layout -> LDS -> two A-fragments (keys 0..31 / 32..63) ----
#pragma unroll
    for (int t = 0; t < 4; ++t)
#pragma unroll
      for (int r = 0; r < 8; ++r)
        ldsP[wave][r + 8 * lhalf][t * 16 + lrow] = (__bf16)pv[t][r];

    BF16x16 pf[2];
#pragma unroll
    for (int s = 0; s < 2; ++s) {
      const __bf16* p = &ldsP[wave][lrow][s * 32 + lhalf * 16];
      pf[s].h[0] = *(const bf16x8*)(p);
      pf[s].h[1] = *(const bf16x8*)(p + 8);
    }

    // ---- row-sum via WMMA: (P x ones) replicated across columns ----
    f32x8 rs = wmma_bf16(pf[0].v, onesf.v, zero);
    rs = wmma_bf16(pf[1].v, onesf.v, rs);
#pragma unroll
    for (int r = 0; r < 8; ++r) l_i[r] = l_i[r] * esc[r] + rs[r];

    // ---- acc = acc * esc + P * V ----
#pragma unroll
    for (int td = 0; td < 4; ++td) {
#pragma unroll
      for (int r = 0; r < 8; ++r) acc[td][r] *= esc[r];
#pragma unroll
      for (int s = 0; s < 2; ++s) {
        BF16x16 vf;  // B[key,d]: contiguous along keys in the Vt tile row
        const __bf16* p = &ldsV[cur][td * 16 + lrow][s * 32 + lhalf * 16];
        vf.h[0] = *(const bf16x8*)(p);
        vf.h[1] = *(const bf16x8*)(p + 8);
        acc[td] = wmma_bf16(pf[s].v, vf.v, acc[td]);
      }
    }
  }

  // ---- normalize + store attn [B, N, H*Hd] bf16 ----
  const int b = bh >> 4, h = bh & 15;
#pragma unroll
  for (int r = 0; r < 8; ++r) {
    const float inv = 1.0f / l_i[r];
    const int n = q0 + r + 8 * lhalf;
    __bf16* orow = attn + ((size_t)(b * NSEQ + n)) * DMODEL + h * HD;
#pragma unroll
    for (int td = 0; td < 4; ++td)
      orow[td * 16 + lrow] = (__bf16)(acc[td][r] * inv);
  }
}

// ======================================================================
extern "C" void kernel_launch(void* const* d_in, const int* in_sizes, int n_in,
                              void* d_out, int out_size, void* d_ws, size_t ws_size,
                              hipStream_t stream) {
  const float* x  = (const float*)d_in[0];
  const float* Wq = (const float*)d_in[1];
  const float* bq = (const float*)d_in[2];
  const float* Wk = (const float*)d_in[3];
  const float* bk = (const float*)d_in[4];
  const float* Wv = (const float*)d_in[5];
  const float* bv = (const float*)d_in[6];
  const float* Wo = (const float*)d_in[7];
  const float* bo = (const float*)d_in[8];

  char* ws = (char*)d_ws;
  const size_t MB = (size_t)1 << 20;
  __bf16* xb   = (__bf16*)(ws + 0 * MB);   // 8 MB  [4096,1024]
  __bf16* wqb  = (__bf16*)(ws + 8 * MB);   // 2 MB
  __bf16* wkb  = (__bf16*)(ws + 10 * MB);  // 2 MB
  __bf16* wvb  = (__bf16*)(ws + 12 * MB);  // 2 MB
  __bf16* wob  = (__bf16*)(ws + 14 * MB);  // 2 MB
  __bf16* Qb   = (__bf16*)(ws + 16 * MB);  // 8 MB  [B,H,N,Hd] (pre-scaled)
  __bf16* Kb   = (__bf16*)(ws + 24 * MB);  // 8 MB  [B,H,N,Hd]
  __bf16* Vtb  = (__bf16*)(ws + 32 * MB);  // 8 MB  [B,H,Hd,N]
  __bf16* attb = (__bf16*)(ws + 40 * MB);  // 8 MB  [B*N, D]

  const int NX = MTOT * DMODEL;
  const int NW = DMODEL * DMODEL;
  cvt_f32_to_bf16<<<NX / 256, 256, 0, stream>>>(x,  xb,  NX);
  cvt_f32_to_bf16<<<NW / 256, 256, 0, stream>>>(Wq, wqb, NW);
  cvt_f32_to_bf16<<<NW / 256, 256, 0, stream>>>(Wk, wkb, NW);
  cvt_f32_to_bf16<<<NW / 256, 256, 0, stream>>>(Wv, wvb, NW);
  cvt_f32_to_bf16<<<NW / 256, 256, 0, stream>>>(Wo, wob, NW);

  proj_gemm<0><<<512, 256, 0, stream>>>(xb, wqb, bq, (void*)Qb, 0.125f);
  proj_gemm<0><<<512, 256, 0, stream>>>(xb, wkb, bk, (void*)Kb, 1.0f);
  proj_gemm<1><<<512, 256, 0, stream>>>(xb, wvb, bv, (void*)Vtb, 1.0f);

  flash_attn<<<512, 256, 0, stream>>>(Qb, Kb, Vtb, attb);

  proj_gemm<2><<<512, 256, 0, stream>>>(attb, wob, bo, d_out, 1.0f);
}
